// SparseUNet3D_4767413698666
// MI455X (gfx1250) — matrix operand order repaired and checked by reading, not compiled, not run on hardware
//
#include <hip/hip_runtime.h>
#include <hip/hip_bf16.h>
#include <stdint.h>

// ---------------------------------------------------------------------------
// SparseUNet3D for MI455X (gfx1250): implicit-GEMM convs via v_wmma_f32_16x16x32_f16
// Activations: fp16, [voxel][channel] (channel-contiguous, mult of 16)
// Weights:     fp16, [tap][co][ci]   (ci padded to mult of 32)
// Per wave: 32 co x 32 voxels (2x2 blocking, 4 accumulators).
// Per block (4 waves): 32co x CI weight tile DMA'd into LDS per tap via
// global_load_async_to_lds_b128 (ASYNCcnt), DOUBLE-BUFFERED so the DMA of
// tap t+1 overlaps the WMMA stream of tap t; in-order async completion lets
// s_wait_asynccnt <= CH stand in for "previous buffer fully landed".
// ---------------------------------------------------------------------------

typedef _Float16 v16h __attribute__((ext_vector_type(16)));
typedef _Float16 v8h  __attribute__((ext_vector_type(8)));
typedef float    v8f  __attribute__((ext_vector_type(8)));

template <int N>
__device__ __forceinline__ void wait_async()
{
    if constexpr (N == 0)      asm volatile("s_wait_asynccnt 0x0" ::: "memory");
    else if constexpr (N == 1) asm volatile("s_wait_asynccnt 0x1" ::: "memory");
    else if constexpr (N == 2) asm volatile("s_wait_asynccnt 0x2" ::: "memory");
    else if constexpr (N == 4) asm volatile("s_wait_asynccnt 0x4" ::: "memory");
    else                       asm volatile("s_wait_asynccnt 0x8" ::: "memory");
}

// MODE: 0 = SubM conv k=3 SAME, 1 = down conv k=2 s=2, 2 = inverse (transpose) conv k=2 s=2
template <int MODE, int CI>
__global__ __launch_bounds__(128)
void conv_wmma(const _Float16* __restrict__ act,
               const _Float16* __restrict__ wpk,
               const float* __restrict__ gamma,
               const float* __restrict__ beta,
               const float* __restrict__ mask,
               _Float16* __restrict__ out,
               int inD, int inH, int inW,
               int outD, int outH, int outW,
               int CO, int ldOut, int coOff)
{
    constexpr int SP = CI + 8;               // padded LDS row stride (halves; 2*SP bytes, 16B mult)
    constexpr int CH = CI / 32;              // async chunks per thread per stage (16B each)
    __shared__ __attribute__((aligned(32))) _Float16 sA[2][32 * SP]; // double-buffered A tiles

    const int tid  = threadIdx.x;
    const int lane = tid & 31;
    const int wave = tid >> 5;
    const int nlo  = lane & 15;              // A: M (co) / B,D: N (voxel)
    const int hi   = lane >> 4;              // half-wave selector
    const int nVox = outD * outH * outW;
    const int coBase = blockIdx.y * 32;

    // two 16-voxel sub-tiles per wave
    const int vb0 = (blockIdx.x * 4 + wave) * 32;
    int oz[2], oy[2], ox[2], vcv[2];
    bool vval[2];
#pragma unroll
    for (int j = 0; j < 2; ++j) {
        int v = vb0 + j * 16 + nlo;
        vval[j] = (v < nVox);
        int vc = vval[j] ? v : 0;
        vcv[j] = vc;
        oz[j] = vc / (outH * outW);
        int rem = vc - oz[j] * (outH * outW);
        oy[j] = rem / outW;
        ox[j] = rem - oy[j] * outW;
    }

    v8f zero = {};
    v8f acc[2][2];
#pragma unroll
    for (int i = 0; i < 2; ++i)
#pragma unroll
        for (int j = 0; j < 2; ++j) acc[i][j] = zero;

    const int NT = (MODE == 0) ? 27 : 8;

    // issue the async DMA of one 32co x CI weight tile into LDS buffer `buf`
    auto issueStage = [&](int tap, int buf) {
        const _Float16* __restrict__ gsrc = wpk + ((size_t)tap * CO + coBase) * CI;
        constexpr int CPR = CI / 8;          // 16B chunks per row; 4*CI chunks (mult of 128)
#pragma unroll 1
        for (int c = tid; c < 4 * CI; c += 128) {
            int row = c / CPR;
            int col = (c - row * CPR) * 8;
            unsigned ldsOff = (unsigned)(uintptr_t)(&sA[buf][row * SP + col]);
            unsigned long long ga =
                (unsigned long long)(uintptr_t)(gsrc + (size_t)row * CI + col);
            asm volatile("global_load_async_to_lds_b128 %0, %1, off"
                         :: "v"(ldsOff), "v"(ga)
                         : "memory");
        }
    };

    issueStage(0, 0);                        // prologue: DMA tap 0 into buffer 0

    for (int tap = 0; tap < NT; ++tap) {
        const int cur = tap & 1;
        __syncthreads();                     // all waves done reading buf[cur^1] (prev tap)
        if (tap + 1 < NT) {
            issueStage(tap + 1, cur ^ 1);    // overlap next tap's DMA with this tap's WMMAs
            wait_async<CH>();                // in-order => tap's own DMA (older) has landed
        } else {
            wait_async<0>();
        }
        __syncthreads();                     // buf[cur] visible to all waves

        // ---- per-sub-tile input geometry for this tap ----
        bool ok[2];
        size_t ivox[2];
#pragma unroll
        for (int j = 0; j < 2; ++j) {
            int iz, iy, ix;
            bool o = vval[j];
            if (MODE == 0) {
                iz = oz[j] + (tap / 9) - 1;
                iy = oy[j] + ((tap / 3) % 3) - 1;
                ix = ox[j] + (tap % 3) - 1;
                o = o && ((unsigned)iz < (unsigned)inD) &&
                         ((unsigned)iy < (unsigned)inH) &&
                         ((unsigned)ix < (unsigned)inW);
            } else if (MODE == 1) {
                iz = 2 * oz[j] + ((tap >> 2) & 1);
                iy = 2 * oy[j] + ((tap >> 1) & 1);
                ix = 2 * ox[j] + (tap & 1);
            } else { // inverse conv: one tap per output-voxel parity
                iz = oz[j] >> 1; iy = oy[j] >> 1; ix = ox[j] >> 1;
                o = o && ((oz[j] & 1) == ((tap >> 2) & 1)) &&
                         ((oy[j] & 1) == ((tap >> 1) & 1)) &&
                         ((ox[j] & 1) == (tap & 1));
            }
            ok[j] = o;
            ivox[j] = o ? ((size_t)(iz * inH + iy) * inW + ix) : 0;
        }
        const _Float16* __restrict__ b0p = act + ivox[0] * CI + hi * 16;
        const _Float16* __restrict__ b1p = act + ivox[1] * CI + hi * 16;
        const _Float16* __restrict__ sAc = &sA[cur][0];

        // ---- K loop (fully unrolled; CI compile-time) ----
#pragma unroll
        for (int kc = 0; kc < CI; kc += 32) {
            // B: 32x16; lane holds 16 contiguous channels of its voxel
            v16h bm0 = {};
            if (ok[0]) bm0 = *(const v16h*)(b0p + kc);
            v16h bm1 = {};
            if (ok[1]) bm1 = *(const v16h*)(b1p + kc);
            // A: 16x32 from LDS; lane holds K run [16*hi,+16)
            const _Float16* s0 = &sAc[nlo * SP + kc + hi * 16];
            const _Float16* s1 = &sAc[(16 + nlo) * SP + kc + hi * 16];
            v8h a00 = *(const v8h*)(s0);
            v8h a01 = *(const v8h*)(s0 + 8);
            v8h a10 = *(const v8h*)(s1);
            v8h a11 = *(const v8h*)(s1 + 8);
            v16h am0 = __builtin_shufflevector(a00, a01,
                0, 1, 2, 3, 4, 5, 6, 7, 8, 9, 10, 11, 12, 13, 14, 15);
            v16h am1 = __builtin_shufflevector(a10, a11,
                0, 1, 2, 3, 4, 5, 6, 7, 8, 9, 10, 11, 12, 13, 14, 15);
            acc[0][0] = __builtin_amdgcn_wmma_f32_16x16x32_f16(false, am0, false, bm0, (short)0, acc[0][0], false, false);
            acc[0][1] = __builtin_amdgcn_wmma_f32_16x16x32_f16(false, am0, false, bm1, (short)0, acc[0][1], false, false);
            acc[1][0] = __builtin_amdgcn_wmma_f32_16x16x32_f16(false, am1, false, bm0, (short)0, acc[1][0], false, false);
            acc[1][1] = __builtin_amdgcn_wmma_f32_16x16x32_f16(false, am1, false, bm1, (short)0, acc[1][1], false, false);
        }
    }

    // ---- epilogue: BN + ReLU + mask, fp16 store ----
#pragma unroll
    for (int j = 0; j < 2; ++j) {
        if (!vval[j]) continue;
        const float mval = mask[vcv[j]];
#pragma unroll
        for (int i = 0; i < 2; ++i) {
            const int co0 = coBase + i * 16 + hi * 8;
            v8h res;
#pragma unroll
            for (int r = 0; r < 8; ++r) {
                float y = acc[i][j][r];
                if (MODE != 2) {
                    y = y * gamma[co0 + r] + beta[co0 + r];
                    y = fmaxf(y, 0.0f);
                }
                res[r] = (_Float16)(y * mval);
            }
            *(v8h*)(out + (size_t)vcv[j] * ldOut + coOff + co0) = res;
        }
    }
}

// x fp32 [CI][nVox] -> fp16 [nVox][CIP] (zero-pad channels)
__global__ void cvt_in_kernel(const float* __restrict__ x, _Float16* __restrict__ o,
                              int nVox, int CI, int CIP)
{
    int v = blockIdx.x * blockDim.x + threadIdx.x;
    if (v >= nVox) return;
    for (int c = 0; c < CIP; ++c)
        o[(size_t)v * CIP + c] = (c < CI) ? (_Float16)x[(size_t)c * nVox + v]
                                          : (_Float16)0.0f;
}

// w fp32 OIDHW [CO][CI][taps] -> fp16 [tap][co][ciP]
__global__ void pack_w_kernel(const float* __restrict__ w, _Float16* __restrict__ o,
                              int CO, int CI, int CIP, int NT)
{
    int idx = blockIdx.x * blockDim.x + threadIdx.x;
    int total = NT * CO * CIP;
    if (idx >= total) return;
    int ci = idx % CIP;
    int t  = idx / CIP;
    int co = t % CO;
    int tap = t / CO;
    o[idx] = (ci < CI) ? (_Float16)w[((size_t)co * CI + ci) * NT + tap]
                       : (_Float16)0.0f;
}

// 2x2x2 max-pool of dense fp32 mask
__global__ void pool_mask_kernel(const float* __restrict__ in, float* __restrict__ o,
                                 int oD, int oH, int oW)
{
    int idx = blockIdx.x * blockDim.x + threadIdx.x;
    int nv = oD * oH * oW;
    if (idx >= nv) return;
    int oz = idx / (oH * oW);
    int r  = idx - oz * oH * oW;
    int oy = r / oW;
    int ox = r - oy * oW;
    int iH = 2 * oH, iW = 2 * oW;
    float m = 0.0f;
    for (int dz = 0; dz < 2; ++dz)
        for (int dy = 0; dy < 2; ++dy)
            for (int dx = 0; dx < 2; ++dx)
                m = fmaxf(m, in[((size_t)(2 * oz + dz) * iH + (2 * oy + dy)) * iW + (2 * ox + dx)]);
    o[idx] = m;
}

// skip-connection copy into concat buffer: d[v*ldD + off + c] = s[v*C + c]
__global__ void copy_skip_kernel(const _Float16* __restrict__ s, _Float16* __restrict__ d,
                                 int C, int ldD, int off, int nVox)
{
    int idx = blockIdx.x * blockDim.x + threadIdx.x;
    int total = nVox * C;
    if (idx >= total) return;
    int v = idx / C, c = idx - v * C;
    d[(size_t)v * ldD + off + c] = s[idx];
}

// head: 1x1x1 conv fp16 -> fp32 NCDHW, + bias, * mask
__global__ void head_kernel(const _Float16* __restrict__ a, const float* __restrict__ hw,
                            const float* __restrict__ hb, const float* __restrict__ mask,
                            float* __restrict__ out, int nVox, int CI, int CO)
{
    int v = blockIdx.x * blockDim.x + threadIdx.x;
    if (v >= nVox) return;
    float m = mask[v];
    for (int co = 0; co < CO; ++co) {
        float s = hb[co];
        for (int ci = 0; ci < CI; ++ci)
            s += (float)a[(size_t)v * CI + ci] * hw[co * CI + ci];
        out[(size_t)co * nVox + v] = s * m;
    }
}

extern "C" void kernel_launch(void* const* d_in, const int* in_sizes, int n_in,
                              void* d_out, int out_size, void* d_ws, size_t ws_size,
                              hipStream_t stream)
{
    (void)in_sizes; (void)n_in; (void)out_size; (void)ws_size;

    const int nv0 = 32 * 96 * 96;   // 294912
    const int nv1 = nv0 / 8;        // 36864
    const int nv2 = nv1 / 8;        // 4608
    const int nv3 = nv2 / 8;        // 576

    auto F = [&](int i) { return (const float*)d_in[i]; };
    const float* x     = F(0);
    const float* mask0 = F(78);

    // bump allocator over workspace
    char* wsp = (char*)d_ws;
    size_t used = 0;
    auto allocB = [&](size_t bytes) -> void* {
        size_t a = (used + 255) & ~(size_t)255;
        void* p = wsp + a;
        used = a + bytes;
        return p;
    };
    auto allocH = [&](size_t halves) -> _Float16* { return (_Float16*)allocB(halves * 2); };
    auto allocF = [&](size_t n) -> float* { return (float*)allocB(n * 4); };

    // activations
    _Float16* X0   = allocH((size_t)nv0 * 32);
    _Float16* B1   = allocH((size_t)nv0 * 32);
    _Float16* B2   = allocH((size_t)nv0 * 32);
    _Float16* CAT0 = allocH((size_t)nv0 * 64);
    _Float16* D0b  = allocH((size_t)nv1 * 64);
    _Float16* T1a  = allocH((size_t)nv1 * 64);
    _Float16* T1b  = allocH((size_t)nv1 * 64);
    _Float16* CAT1 = allocH((size_t)nv1 * 128);
    _Float16* D1b  = allocH((size_t)nv2 * 128);
    _Float16* T2a  = allocH((size_t)nv2 * 128);
    _Float16* T2b  = allocH((size_t)nv2 * 128);
    _Float16* CAT2 = allocH((size_t)nv2 * 256);
    _Float16* D2b  = allocH((size_t)nv3 * 256);
    _Float16* T3a  = allocH((size_t)nv3 * 256);
    _Float16* T3b  = allocH((size_t)nv3 * 256);
    float* m1 = allocF(nv1);
    float* m2 = allocF(nv2);
    float* m3 = allocF(nv3);

    auto pack = [&](int widx, int CO, int CI, int CIP, int NT) -> const _Float16* {
        _Float16* p = allocH((size_t)NT * CO * CIP);
        int total = NT * CO * CIP;
        pack_w_kernel<<<(total + 255) / 256, 256, 0, stream>>>(F(widx), p, CO, CI, CIP, NT);
        return p;
    };

    // pack all weights fp32 -> fp16 [tap][co][ci]
    const _Float16* we0[3]  = { pack(1, 32, 20, 32, 27),  pack(4, 32, 32, 32, 27),  pack(7, 32, 32, 32, 27) };
    const _Float16* wd0     =   pack(10, 64, 32, 32, 8);
    const _Float16* we1[3]  = { pack(13, 64, 64, 64, 27), pack(16, 64, 64, 64, 27), pack(19, 64, 64, 64, 27) };
    const _Float16* wd1     =   pack(22, 128, 64, 64, 8);
    const _Float16* we2[3]  = { pack(25, 128, 128, 128, 27), pack(28, 128, 128, 128, 27), pack(31, 128, 128, 128, 27) };
    const _Float16* wd2     =   pack(34, 256, 128, 128, 8);
    const _Float16* wb[3]   = { pack(37, 256, 256, 256, 27), pack(40, 256, 256, 256, 27), pack(43, 256, 256, 256, 27) };
    const _Float16* wu2     =   pack(46, 128, 256, 256, 8);
    const _Float16* wdc2[3] = { pack(47, 128, 256, 256, 27), pack(50, 128, 128, 128, 27), pack(53, 128, 128, 128, 27) };
    const _Float16* wu1     =   pack(56, 64, 128, 128, 8);
    const _Float16* wdc1[3] = { pack(57, 64, 128, 128, 27), pack(60, 64, 64, 64, 27), pack(63, 64, 64, 64, 27) };
    const _Float16* wu0     =   pack(66, 32, 64, 64, 8);
    const _Float16* wdc0[3] = { pack(67, 32, 64, 64, 27), pack(70, 32, 32, 32, 27), pack(73, 32, 32, 32, 27) };

    // pooled masks
    pool_mask_kernel<<<(nv1 + 255) / 256, 256, 0, stream>>>(mask0, m1, 16, 48, 48);
    pool_mask_kernel<<<(nv2 + 255) / 256, 256, 0, stream>>>(m1, m2, 8, 24, 24);
    pool_mask_kernel<<<(nv3 + 255) / 256, 256, 0, stream>>>(m2, m3, 4, 12, 12);

    // input fp32 [20][vox] -> fp16 [vox][32]
    cvt_in_kernel<<<(nv0 + 255) / 256, 256, 0, stream>>>(x, X0, nv0, 20, 32);

    auto conv = [&](int mode, const _Float16* in, const _Float16* wp,
                    const float* g, const float* b, const float* m,
                    _Float16* o, int iD, int iH, int iW, int oD, int oH, int oW,
                    int CIv, int CO, int ldOut, int coOff) {
        int nVox = oD * oH * oW;
        dim3 grid((nVox + 127) / 128, CO / 32);
        dim3 blk(128);
#define LAUNCH_CONV(M, C) conv_wmma<M, C><<<grid, blk, 0, stream>>>( \
            in, wp, g, b, m, o, iD, iH, iW, oD, oH, oW, CO, ldOut, coOff)
        if (mode == 0) {
            if (CIv == 32) LAUNCH_CONV(0, 32);
            else if (CIv == 64) LAUNCH_CONV(0, 64);
            else if (CIv == 128) LAUNCH_CONV(0, 128);
            else LAUNCH_CONV(0, 256);
        } else if (mode == 1) {
            if (CIv == 32) LAUNCH_CONV(1, 32);
            else if (CIv == 64) LAUNCH_CONV(1, 64);
            else LAUNCH_CONV(1, 128);
        } else {
            if (CIv == 64) LAUNCH_CONV(2, 64);
            else if (CIv == 128) LAUNCH_CONV(2, 128);
            else LAUNCH_CONV(2, 256);
        }
#undef LAUNCH_CONV
    };

    // ---- encoder level 0 ----
    conv(0, X0, we0[0], F(2), F(3), mask0, B1, 32, 96, 96, 32, 96, 96, 32, 32, 32, 0);
    conv(0, B1, we0[1], F(5), F(6), mask0, B2, 32, 96, 96, 32, 96, 96, 32, 32, 32, 0);
    conv(0, B2, we0[2], F(8), F(9), mask0, X0, 32, 96, 96, 32, 96, 96, 32, 32, 32, 0); // e0 = X0
    conv(1, X0, wd0, F(11), F(12), m1, D0b, 32, 96, 96, 16, 48, 48, 32, 64, 64, 0);
    // ---- encoder level 1 ----
    conv(0, D0b, we1[0], F(14), F(15), m1, T1a, 16, 48, 48, 16, 48, 48, 64, 64, 64, 0);
    conv(0, T1a, we1[1], F(17), F(18), m1, T1b, 16, 48, 48, 16, 48, 48, 64, 64, 64, 0);
    conv(0, T1b, we1[2], F(20), F(21), m1, T1a, 16, 48, 48, 16, 48, 48, 64, 64, 64, 0); // e1 = T1a
    conv(1, T1a, wd1, F(23), F(24), m2, D1b, 16, 48, 48, 8, 24, 24, 64, 128, 128, 0);
    // ---- encoder level 2 ----
    conv(0, D1b, we2[0], F(26), F(27), m2, T2a, 8, 24, 24, 8, 24, 24, 128, 128, 128, 0);
    conv(0, T2a, we2[1], F(29), F(30), m2, T2b, 8, 24, 24, 8, 24, 24, 128, 128, 128, 0);
    conv(0, T2b, we2[2], F(32), F(33), m2, T2a, 8, 24, 24, 8, 24, 24, 128, 128, 128, 0); // e2 = T2a
    conv(1, T2a, wd2, F(35), F(36), m3, D2b, 8, 24, 24, 4, 12, 12, 128, 256, 256, 0);
    // ---- bottleneck ----
    conv(0, D2b, wb[0], F(38), F(39), m3, T3a, 4, 12, 12, 4, 12, 12, 256, 256, 256, 0);
    conv(0, T3a, wb[1], F(41), F(42), m3, T3b, 4, 12, 12, 4, 12, 12, 256, 256, 256, 0);
    conv(0, T3b, wb[2], F(44), F(45), m3, T3a, 4, 12, 12, 4, 12, 12, 256, 256, 256, 0); // b = T3a
    // ---- decoder level 2 ----
    conv(2, T3a, wu2, nullptr, nullptr, m2, CAT2, 4, 12, 12, 8, 24, 24, 256, 128, 256, 0);
    copy_skip_kernel<<<((size_t)nv2 * 128 + 255) / 256, 256, 0, stream>>>(T2a, CAT2, 128, 256, 128, nv2);
    conv(0, CAT2, wdc2[0], F(48), F(49), m2, T2b, 8, 24, 24, 8, 24, 24, 256, 128, 128, 0);
    conv(0, T2b, wdc2[1], F(51), F(52), m2, T2a, 8, 24, 24, 8, 24, 24, 128, 128, 128, 0);
    conv(0, T2a, wdc2[2], F(54), F(55), m2, T2b, 8, 24, 24, 8, 24, 24, 128, 128, 128, 0); // c2 = T2b
    // ---- decoder level 1 ----
    conv(2, T2b, wu1, nullptr, nullptr, m1, CAT1, 8, 24, 24, 16, 48, 48, 128, 64, 128, 0);
    copy_skip_kernel<<<((size_t)nv1 * 64 + 255) / 256, 256, 0, stream>>>(T1a, CAT1, 64, 128, 64, nv1);
    conv(0, CAT1, wdc1[0], F(58), F(59), m1, T1b, 16, 48, 48, 16, 48, 48, 128, 64, 64, 0);
    conv(0, T1b, wdc1[1], F(61), F(62), m1, T1a, 16, 48, 48, 16, 48, 48, 64, 64, 64, 0);
    conv(0, T1a, wdc1[2], F(64), F(65), m1, T1b, 16, 48, 48, 16, 48, 48, 64, 64, 64, 0); // c1 = T1b
    // ---- decoder level 0 ----
    conv(2, T1b, wu0, nullptr, nullptr, mask0, CAT0, 16, 48, 48, 32, 96, 96, 64, 32, 64, 0);
    copy_skip_kernel<<<((size_t)nv0 * 32 + 255) / 256, 256, 0, stream>>>(X0, CAT0, 32, 64, 32, nv0);
    conv(0, CAT0, wdc0[0], F(68), F(69), mask0, B1, 32, 96, 96, 32, 96, 96, 64, 32, 32, 0);
    conv(0, B1, wdc0[1], F(71), F(72), mask0, B2, 32, 96, 96, 32, 96, 96, 32, 32, 32, 0);
    conv(0, B2, wdc0[2], F(74), F(75), mask0, B1, 32, 96, 96, 32, 96, 96, 32, 32, 32, 0); // c0 = B1
    // ---- head ----
    head_kernel<<<(nv0 + 255) / 256, 256, 0, stream>>>(B1, F(76), F(77), mask0,
                                                       (float*)d_out, nv0, 32, 25);
}
